// MHA_26087631356345
// MI455X (gfx1250) — compile-verified, matrix operand order checked by reading, not run on hardware
//
#include <hip/hip_runtime.h>
#include <hip/hip_bf16.h>

typedef _Float16 half_t;
typedef __attribute__((ext_vector_type(16))) _Float16 v16h;
typedef __attribute__((ext_vector_type(8)))  float    v8f;
typedef __attribute__((ext_vector_type(4)))  unsigned int v4u;
typedef __attribute__((ext_vector_type(8)))  int      v8i;
typedef __attribute__((ext_vector_type(4)))  int      v4i;

#define D_MODEL 768
#define SEQ     1024
#define BATCH   8
#define NHEADS  12
#define HDIM    64
#define NTOK    (BATCH * SEQ)

#if defined(__has_builtin)
#  if __has_builtin(__builtin_amdgcn_tensor_load_to_lds)
#    define HAVE_TDM 1
#  else
#    define HAVE_TDM 0
#  endif
#else
#  define HAVE_TDM 0
#endif

// ---------------------------------------------------------------------------
// Issue an async TDM load of one 64x64 f16 tile (row stride HDIM) into LDS.
// 2D D# descriptor: data_size=2B, tensor 64x64, tile 64x64, dim0_stride=64.
// No wait here -- pair with stage_wait() + __syncthreads() (double buffering).
// ---------------------------------------------------------------------------
__device__ inline void stage_issue(const half_t* __restrict__ g,
                                   half_t (*lds)[HDIM], int tid, int w) {
#if HAVE_TDM
    if (w == 0) {
        unsigned long long ga = (unsigned long long)(size_t)g;
        unsigned lo = (unsigned)(size_t)(&lds[0][0]);   // LDS aperture: addr[31:0]
        v4u g0;
        g0[0] = 1u;                                     // count=1, user descriptor
        g0[1] = lo;                                     // lds_addr (bytes)
        g0[2] = (unsigned)(ga & 0xffffffffu);           // global_addr[31:0]
        g0[3] = (unsigned)((ga >> 32) & 0x1ffffffu)     // global_addr[56:32]
              | (2u << 30);                             // type=2 ("image")
        v8i g1;
        g1[0] = (1 << 16);      // workgroup_mask=0, data_size=1 (2 bytes)
        g1[1] = (64 << 16);     // tensor_dim0[15:0]=64 at bits 63:48
        g1[2] = (64 << 16);     // tensor_dim0 hi=0; tensor_dim1[15:0]=64
        g1[3] = (64 << 16);     // tensor_dim1 hi=0; tile_dim0=64
        g1[4] = 64;             // tile_dim1=64; tile_dim2=0
        g1[5] = 64;             // tensor_dim0_stride[31:0]=64 elems
        g1[6] = 0;
        g1[7] = 0;
        v4i g2 = {0, 0, 0, 0};  // 2D: groups 2/3 unused
        v4i g3 = {0, 0, 0, 0};
        v8i g4 = {0, 0, 0, 0, 0, 0, 0, 0};
        __builtin_amdgcn_tensor_load_to_lds(g0, g1, g2, g3, g4, 0);
    }
#else
    int base = tid * 16;
    int i = base >> 6, j = base & 63;
    #pragma unroll
    for (int u = 0; u < 16; ++u)
        lds[i][j + u] = g[(size_t)i * HDIM + j + u];
#endif
}

__device__ inline void stage_wait(int w) {
#if HAVE_TDM
    if (w == 0) __builtin_amdgcn_s_wait_tensorcnt((short)0);
#endif
}

// ---------------------------------------------------------------------------
// Kernel 1: fused QKV projection.  out = x @ W^T + b, cast to f16, laid out
// [B, H, S, Dh].  Q pre-scaled by 1/sqrt(Dh).  Double-buffered LDS staging:
// next tile's global loads fly while WMMA runs on the current buffer;
// a single barrier per K-step seals the swap.
// ---------------------------------------------------------------------------
__global__ __launch_bounds__(256) void qkv_proj_kernel(
    const float* __restrict__ x,
    const float* __restrict__ Wq, const float* __restrict__ bq,
    const float* __restrict__ Wk, const float* __restrict__ bk,
    const float* __restrict__ Wv, const float* __restrict__ bv,
    half_t* __restrict__ qh, half_t* __restrict__ kh, half_t* __restrict__ vh)
{
    __shared__ half_t As[2][64][32];
    __shared__ half_t Bs[2][64][32];

    const int proj = blockIdx.z;
    const float* W    = (proj == 0) ? Wq : (proj == 1) ? Wk : Wv;
    const float* bias = (proj == 0) ? bq : (proj == 1) ? bk : bv;
    half_t* dst       = (proj == 0) ? qh : (proj == 1) ? kh : vh;
    const float outscale = (proj == 0) ? 0.125f : 1.0f;

    const int row0  = blockIdx.x * 64;
    const int feat0 = blockIdx.y * 64;
    const int head  = blockIdx.y;

    const int tid  = threadIdx.x;
    const int lane = tid & 31;
    const int w    = tid >> 5;
    const int wm   = w & 3;
    const int wn   = w >> 2;

    const int sbase = tid * 8;
    const int si = sbase >> 5, sj = sbase & 31;
    const float* xrow = &x[(size_t)(row0  + si) * D_MODEL + sj];
    const float* wrow = &W[(size_t)(feat0 + si) * D_MODEL + sj];

    {   // prologue: stage K-tile 0 into buffer 0
        #pragma unroll
        for (int u = 0; u < 8; ++u) {
            As[0][si][sj + u] = (half_t)xrow[u];
            Bs[0][si][sj + u] = (half_t)wrow[u];
        }
        __syncthreads();
    }

    v8f c0 = {}, c1 = {};
    const int m  = wm * 16 + (lane & 15);
    const int n0 = wn * 32 + (lane & 15);
    const int n1 = n0 + 16;

    int p = 0;
    const int NIT = D_MODEL / 32;
    for (int it = 0; it < NIT; ++it) {
        const int knext = (it + 1) * 32;
        float xa[8], wa[8];
        if (it + 1 < NIT) {          // issue next tile's global loads early
            if (it + 2 < NIT) __builtin_prefetch(&xrow[knext + 32], 0, 1);
            #pragma unroll
            for (int u = 0; u < 8; ++u) { xa[u] = xrow[knext + u]; wa[u] = wrow[knext + u]; }
        }

        v16h a, b0, b1;                       // compute on buffer p
        #pragma unroll
        for (int j = 0; j < 8; ++j) {
            int kk = ((j & 4) << 2) + ((lane & 16) >> 1) + ((j & 3) << 1);
            a [2*j] = As[p][m ][kk]; a [2*j+1] = As[p][m ][kk+1];
            b0[2*j] = Bs[p][n0][kk]; b0[2*j+1] = Bs[p][n0][kk+1];
            b1[2*j] = Bs[p][n1][kk]; b1[2*j+1] = Bs[p][n1][kk+1];
        }
        c0 = __builtin_amdgcn_wmma_f32_16x16x32_f16(false, a, false, b0, (short)0, c0, false, false);
        c1 = __builtin_amdgcn_wmma_f32_16x16x32_f16(false, a, false, b1, (short)0, c1, false, false);

        if (it + 1 < NIT) {          // convert + store into spare buffer
            #pragma unroll
            for (int u = 0; u < 8; ++u) {
                As[p ^ 1][si][sj + u] = (half_t)xa[u];
                Bs[p ^ 1][si][sj + u] = (half_t)wa[u];
            }
        }
        __syncthreads();
        p ^= 1;
    }

    #pragma unroll
    for (int acc = 0; acc < 2; ++acc) {
        const v8f& c = acc ? c1 : c0;
        int dloc = wn * 32 + acc * 16 + (lane & 15);
        float bb = bias[feat0 + dloc];
        #pragma unroll
        for (int r = 0; r < 8; ++r) {
            int t  = row0 + wm * 16 + r + ((lane & 16) >> 1);
            int bi = t >> 10, si2 = t & 1023;
            size_t di = (((size_t)bi * NHEADS + head) * SEQ + si2) * HDIM + dloc;
            dst[di] = (half_t)((c[r] + bb) * outscale);
        }
    }
}

// ---------------------------------------------------------------------------
// Kernel 2: fused attention for one (head, 64-query block).
// Phase A: S = Q K^T -> 64x1024 f32 strip in LDS (256 KB, CDNA5-only)
// Phase B: wave-per-8-rows softmax with shfl_xor reductions; probs streamed
//          to HBM once as 512B-contiguous wave stores (the roofline term)
// Phase C: H = P V, P fragments converted f32->f16 straight from LDS
// K/V tiles: double-buffered Tensor Data Mover loads overlapped with WMMA.
// ---------------------------------------------------------------------------
__global__ __launch_bounds__(256) void attn_kernel(
    const half_t* __restrict__ qh, const half_t* __restrict__ kh,
    const half_t* __restrict__ vh,
    float* __restrict__ out_h, float* __restrict__ out_p)
{
    __shared__ float  Sblk[64][1024];      // 256 KB score strip
    __shared__ half_t Qs[64][HDIM];        // 8 KB
    __shared__ half_t KVs[2][64][HDIM];    // 16 KB double-buffered K/V tiles

    const int qt = blockIdx.x;
    const int bh = blockIdx.y;
    const int bi = bh / NHEADS, hi = bh % NHEADS;
    const int q0 = qt * 64;
    const size_t head_off = (size_t)bh * SEQ * HDIM;
    const half_t* Q = qh + head_off;
    const half_t* K = kh + head_off;
    const half_t* V = vh + head_off;

    const int tid  = threadIdx.x;
    const int lane = tid & 31;
    const int w    = tid >> 5;
    const int wm   = w & 3;
    const int wn   = w >> 2;
    const int m    = wm * 16 + (lane & 15);

    // ---- Phase A: scores into LDS (TDM double-buffered over K tiles) ----
    stage_issue(Q + (size_t)q0 * HDIM, Qs, tid, w);
    stage_issue(K, KVs[0], tid, w);
    stage_wait(w);
    __syncthreads();

    int cur = 0;
    for (int kt = 0; kt < SEQ; kt += 64) {
        if (kt + 64 < SEQ)
            stage_issue(K + (size_t)(kt + 64) * HDIM, KVs[cur ^ 1], tid, w);
        #pragma unroll
        for (int acc = 0; acc < 2; ++acc) {
            const int nk = wn * 32 + acc * 16 + (lane & 15);
            v8f c = {};
            #pragma unroll
            for (int dch = 0; dch < 2; ++dch) {
                v16h a, b;
                #pragma unroll
                for (int j = 0; j < 8; ++j) {
                    int kk = dch * 32 + ((j & 4) << 2) + ((lane & 16) >> 1) + ((j & 3) << 1);
                    a[2*j] = Qs[m][kk];        a[2*j+1] = Qs[m][kk+1];
                    b[2*j] = KVs[cur][nk][kk]; b[2*j+1] = KVs[cur][nk][kk+1];
                }
                c = __builtin_amdgcn_wmma_f32_16x16x32_f16(false, a, false, b, (short)0, c, false, false);
            }
            #pragma unroll
            for (int r = 0; r < 8; ++r)
                Sblk[wm * 16 + r + ((lane & 16) >> 1)][kt + nk] = c[r];
        }
        stage_wait(w);
        __syncthreads();
        cur ^= 1;
    }

    // ---- Phase B: softmax, one wave per 8 rows, coalesced probs stream ----
    {
        #pragma unroll 1
        for (int rr = 0; rr < 8; ++rr) {
            const int r = w * 8 + rr;
            float* row = Sblk[r];
            float4 vb[8];
            float mx = -3.4e38f;
            #pragma unroll
            for (int i = 0; i < 8; ++i) {
                float4 t = *(float4*)&row[(i * 32 + lane) * 4];
                vb[i] = t;
                mx = fmaxf(mx, fmaxf(fmaxf(t.x, t.y), fmaxf(t.z, t.w)));
            }
            #pragma unroll
            for (int s = 16; s; s >>= 1) mx = fmaxf(mx, __shfl_xor(mx, s, 32));
            float sum = 0.f;
            #pragma unroll
            for (int i = 0; i < 8; ++i) {
                float4 t = vb[i];
                t.x = __expf(t.x - mx); t.y = __expf(t.y - mx);
                t.z = __expf(t.z - mx); t.w = __expf(t.w - mx);
                vb[i] = t;
                sum += (t.x + t.y) + (t.z + t.w);
            }
            #pragma unroll
            for (int s = 16; s; s >>= 1) sum += __shfl_xor(sum, s, 32);
            const float inv = 1.f / sum;
            float* gp = out_p + ((size_t)bh * SEQ + (q0 + r)) * SEQ;
            #pragma unroll
            for (int i = 0; i < 8; ++i) {
                float4 t = vb[i];
                t.x *= inv; t.y *= inv; t.z *= inv; t.w *= inv;
                *(float4*)&row[(i * 32 + lane) * 4] = t;   // probs back to LDS
                *(float4*)&gp [(i * 32 + lane) * 4] = t;   // single HBM write
            }
        }
    }
    __syncthreads();

    // ---- Phase C: H = P @ V (TDM double-buffered over V tiles) ----
    stage_issue(V, KVs[0], tid, w);
    stage_wait(w);
    __syncthreads();

    v8f h0 = {}, h1 = {};
    cur = 0;
    for (int kt = 0; kt < SEQ; kt += 64) {
        if (kt + 64 < SEQ)
            stage_issue(V + (size_t)(kt + 64) * HDIM, KVs[cur ^ 1], tid, w);
        #pragma unroll
        for (int kc = 0; kc < 2; ++kc) {
            v16h a;
            #pragma unroll
            for (int j = 0; j < 8; ++j) {
                int kk = kc * 32 + ((j & 4) << 2) + ((lane & 16) >> 1) + ((j & 3) << 1);
                float2 pp = *(const float2*)&Sblk[m][kt + kc * 0 + kk];
                a[2*j] = (half_t)pp.x; a[2*j+1] = (half_t)pp.y;
            }
            #pragma unroll
            for (int acc = 0; acc < 2; ++acc) {
                const int nd = wn * 32 + acc * 16 + (lane & 15);
                v16h b;
                #pragma unroll
                for (int e = 0; e < 16; ++e) {
                    int kk = kc * 32 + ((e & 8) << 1) + ((lane & 16) >> 1) + (e & 7);
                    b[e] = KVs[cur][kk][nd];
                }
                if (acc == 0)
                    h0 = __builtin_amdgcn_wmma_f32_16x16x32_f16(false, a, false, b, (short)0, h0, false, false);
                else
                    h1 = __builtin_amdgcn_wmma_f32_16x16x32_f16(false, a, false, b, (short)0, h1, false, false);
            }
        }
        stage_wait(w);
        __syncthreads();
        cur ^= 1;
    }

    #pragma unroll
    for (int acc = 0; acc < 2; ++acc) {
        const v8f& c = acc ? h1 : h0;
        int nd = wn * 32 + acc * 16 + (lane & 15);
        int f  = hi * HDIM + nd;
        #pragma unroll
        for (int r = 0; r < 8; ++r) {
            int srow = q0 + wm * 16 + r + ((lane & 16) >> 1);
            out_h[((size_t)bi * SEQ + srow) * D_MODEL + f] = c[r];
        }
    }
}

// ---------------------------------------------------------------------------
extern "C" void kernel_launch(void* const* d_in, const int* in_sizes, int n_in,
                              void* d_out, int out_size, void* d_ws, size_t ws_size,
                              hipStream_t stream) {
    const float* x  = (const float*)d_in[0];
    const float* Wq = (const float*)d_in[1];
    const float* bq = (const float*)d_in[2];
    const float* Wk = (const float*)d_in[3];
    const float* bk = (const float*)d_in[4];
    const float* Wv = (const float*)d_in[5];
    const float* bv = (const float*)d_in[6];

    const size_t headsz = (size_t)BATCH * NHEADS * SEQ * HDIM;
    half_t* qh = (half_t*)d_ws;
    half_t* kh = qh + headsz;
    half_t* vh = kh + headsz;

    float* out_h = (float*)d_out;
    float* out_p = out_h + (size_t)BATCH * SEQ * D_MODEL;

    dim3 g1(NTOK / 64, D_MODEL / 64, 3);
    qkv_proj_kernel<<<g1, 256, 0, stream>>>(x, Wq, bq, Wk, bk, Wv, bv, qh, kh, vh);

    dim3 g2(SEQ / 64, BATCH * NHEADS);
    attn_kernel<<<g2, 256, 0, stream>>>(qh, kh, vh, out_h, out_p);
}
